// QLSTMIDS_45002667328035
// MI455X (gfx1250) — 8-bit matrix operand order repaired and checked by reading, compiled, not run on hardware
//
#include <hip/hip_runtime.h>
#include <math.h>

typedef __attribute__((ext_vector_type(8))) int   v8i;
typedef __attribute__((ext_vector_type(8))) float v8f;
typedef __attribute__((ext_vector_type(2))) float v2f;

#define T_STEPS 256
#define NIN   10
#define NHID  20
#define NGATE 80
#define WAVES_PER_BLOCK 8

#define WS_CNT   0
#define WS_GEN   1
#define WS_SLOTS 2
#define WS_MAXHN 18
#define WS_MAXA1 19
#define WS_MAXA2 20
#define WS_XMAX  32
#define WS_HN    288

__device__ __forceinline__ unsigned f2key(float f){
  unsigned u = __float_as_uint(f);
  return (u & 0x80000000u) ? ~u : (u | 0x80000000u);
}
__device__ __forceinline__ float key2f(unsigned k){
  unsigned u = (k & 0x80000000u) ? (k ^ 0x80000000u) : ~k;
  return __uint_as_float(u);
}
__device__ __forceinline__ float wave_max(float v){
  #pragma unroll
  for(int off = 16; off > 0; off >>= 1) v = fmaxf(v, __shfl_xor(v, off, 32));
  return v;
}
__device__ __forceinline__ float sigm(float x){ return 1.f/(1.f + expf(-x)); }

__device__ __forceinline__ void grid_barrier(unsigned* cnt, volatile unsigned* gen, unsigned nb){
  __threadfence();
  __syncthreads();
  if(threadIdx.x == 0){
    unsigned g = *gen;
    if(atomicAdd(cnt, 1u) == nb - 1u){
      *cnt = 0u;
      __threadfence();
      atomicAdd((unsigned*)gen, 1u);
    } else {
      while(*gen == g){ __builtin_amdgcn_s_sleep(1); }
    }
    __threadfence();
  }
  __syncthreads();
}

__device__ __forceinline__ int q8i(float x, float inv_s){
  float r = rintf(x * inv_s);
  r = fminf(fmaxf(r, -128.f), 127.f);
  return ((int)r) & 0xFF;
}

__global__ void init_ws(unsigned* wsu){
  if(threadIdx.x < 32) wsu[threadIdx.x] = 0u;
}

__global__ __launch_bounds__(256) void xmax_kernel(const float* __restrict__ x,
                                                   float* __restrict__ xmax, int B){
  __shared__ float red[256];
  const int t = blockIdx.x;
  float m = 0.f;
  for(int bb = threadIdx.x; bb < B; bb += blockDim.x){
    const float* p = x + ((long)bb * T_STEPS + t) * NIN;
    #pragma unroll
    for(int i = 0; i < NIN; i++) m = fmaxf(m, fabsf(p[i]));
  }
  red[threadIdx.x] = m; __syncthreads();
  for(int s = 128; s > 0; s >>= 1){
    if(threadIdx.x < s) red[threadIdx.x] = fmaxf(red[threadIdx.x], red[threadIdx.x + s]);
    __syncthreads();
  }
  if(threadIdx.x == 0) xmax[t] = red[0];
}

__global__ __launch_bounds__(256) void lstm_scan(
    const float* __restrict__ x, const float* __restrict__ w_ih,
    const float* __restrict__ w_hh, const float* __restrict__ bias,
    float* __restrict__ wsf, unsigned* __restrict__ wsu,
    float* __restrict__ hn, int B, int nb)
{
  __shared__ float sh_h[WAVES_PER_BLOCK*16][NHID];
  __shared__ float sh_c[WAVES_PER_BLOCK*16][NHID];
  __shared__ float sh_g[WAVES_PER_BLOCK][16][NGATE];

  const int  wv   = threadIdx.x >> 5;
  const int  lane = threadIdx.x & 31;
  const int  mrow = lane & 15;
  const int  hi   = lane >> 4;
  const int  lr0  = wv * 16;
  const long rb   = ((long)blockIdx.x * WAVES_PER_BLOCK + wv) * 16;

  for(int e = lane; e < 16*NHID; e += 32){
    sh_h[lr0 + e/NHID][e%NHID] = 0.f;
    sh_c[lr0 + e/NHID][e%NHID] = 0.f;
  }

  float wm = 0.f;
  for(int i = lane; i < NGATE*NIN;  i += 32) wm = fmaxf(wm, fabsf(w_ih[i]));
  const float s_wih = fmaxf(wave_max(wm), 1e-8f) / 127.f;
  const float inv_swih = 1.f / s_wih;
  wm = 0.f;
  for(int i = lane; i < NGATE*NHID; i += 32) wm = fmaxf(wm, fabsf(w_hh[i]));
  const float s_whh = fmaxf(wave_max(wm), 1e-8f) / 127.f;
  const float inv_swhh = 1.f / s_whh;

  v8i Bih[5], Bhh[5];
  float biasn[5];
  #pragma unroll
  for(int nt = 0; nt < 5; nt++){
    const int n = nt*16 + mrow;
    biasn[nt] = bias[n];
    #pragma unroll
    for(int v = 0; v < 8; v++){
      const int kb = ((v >= 4) ? 32 : 0) + (hi ? 16 : 0) + (v & 3)*4;
      unsigned pa = 0u, ph = 0u;
      #pragma unroll
      for(int j = 0; j < 4; j++){
        const int k = kb + j;
        const int qa = (k < NIN ) ? q8i(w_ih[n*NIN  + k], inv_swih) : 0;
        const int qh = (k < NHID) ? q8i(w_hh[n*NHID + k], inv_swhh) : 0;
        pa |= ((unsigned)qa) << (8*j);
        ph |= ((unsigned)qh) << (8*j);
      }
      Bih[nt][v] = (int)pa;
      Bhh[nt][v] = (int)ph;
    }
  }

  unsigned* slots_all = wsu + WS_SLOTS;
  const float* xmax = wsf + WS_XMAX;

  float s_h = 1e-8f/127.f, inv_sh = 127.f/1e-8f;
  float homax_g = 0.f;

  for(int t = 0; t < T_STEPS; t++){
    const int p = t & 1;
    unsigned* slots  = slots_all + p*8;
    unsigned* oslots = slots_all + (1-p)*8;

    const float s_x = fmaxf(xmax[t], 1e-8f) / 127.f;
    const float inv_sx = 1.f / s_x;

    v8i Ax, Ah;
    #pragma unroll
    for(int v = 0; v < 8; v++){
      const int kb = ((v >= 4) ? 32 : 0) + (hi ? 16 : 0) + (v & 3)*4;
      unsigned pa = 0u, ph = 0u;
      #pragma unroll
      for(int j = 0; j < 4; j++){
        const int k = kb + j;
        int qa = 0, qh = 0;
        if(k < NIN)  qa = q8i(x[((rb + mrow)*T_STEPS + t)*NIN + k], inv_sx);
        if(k < NHID) qh = q8i(sh_h[lr0 + mrow][k], inv_sh);
        pa |= ((unsigned)qa) << (8*j);
        ph |= ((unsigned)qh) << (8*j);
      }
      Ax[v] = (int)pa;
      Ah[v] = (int)ph;
    }
    if(t + 1 < T_STEPS)
      __builtin_prefetch(&x[((rb + mrow)*T_STEPS + (t+1))*NIN], 0, 1);

    const float sxw = s_x * s_wih, shw = s_h * s_whh;
    float gmax = 0.f, gim = -3.4e38f, gfm = -3.4e38f, ggm = 0.f, gom = -3.4e38f;
    #pragma unroll
    for(int nt = 0; nt < 5; nt++){
      v8i zero = {0,0,0,0,0,0,0,0};
      v8i ax = __builtin_amdgcn_wmma_i32_16x16x64_iu8(true, Ax, true, Bih[nt], zero, false, false);
      v8i ah = __builtin_amdgcn_wmma_i32_16x16x64_iu8(true, Ah, true, Bhh[nt], zero, false, false);
      const int n = nt*16 + mrow;
      const int gid = n / NHID;
      #pragma unroll
      for(int r = 0; r < 8; r++){
        const float g = sxw * (float)ax[r] + shw * (float)ah[r] + biasn[nt];
        sh_g[wv][r + hi*8][n] = g;
        gmax = fmaxf(gmax, fabsf(g));
        if(gid == 0)      gim = fmaxf(gim, g);
        else if(gid == 1) gfm = fmaxf(gfm, g);
        else if(gid == 2) ggm = fmaxf(ggm, fabsf(g));
        else              gom = fmaxf(gom, g);
      }
    }
    gmax = wave_max(gmax); gim = wave_max(gim); gfm = wave_max(gfm);
    ggm  = wave_max(ggm);  gom = wave_max(gom);
    if(lane == 0){
      atomicMax(&slots[0], f2key(gmax));
      atomicMax(&slots[1], f2key(gim));
      atomicMax(&slots[2], f2key(gfm));
      atomicMax(&slots[3], f2key(ggm));
      atomicMax(&slots[4], f2key(gom));
    }
    grid_barrier(wsu + WS_CNT, wsu + WS_GEN, nb);
    if(blockIdx.x == 0 && threadIdx.x == 0){
      oslots[0]=0u; oslots[1]=0u; oslots[2]=0u; oslots[3]=0u; oslots[4]=0u;
    }
    const float Gm  = key2f(((volatile unsigned*)slots)[0]);
    const float Gim = key2f(((volatile unsigned*)slots)[1]);
    const float Gfm = key2f(((volatile unsigned*)slots)[2]);
    const float Ggm = key2f(((volatile unsigned*)slots)[3]);
    const float Gom = key2f(((volatile unsigned*)slots)[4]);

    const float s_g = fmaxf(Gm, 1e-8f) / 31.f;
    const float inv_sg = 1.f / s_g;
    const float s_i  = fmaxf(sigm(fminf(fmaxf(rintf(Gim*inv_sg),-32.f),31.f)*s_g), 1e-8f)/63.f;
    const float s_f  = fmaxf(sigm(fminf(fmaxf(rintf(Gfm*inv_sg),-32.f),31.f)*s_g), 1e-8f)/63.f;
    const float s_o  = fmaxf(sigm(fminf(fmaxf(rintf(Gom*inv_sg),-32.f),31.f)*s_g), 1e-8f)/63.f;
    const float s_gt = fmaxf(tanhf(fminf(rintf(Ggm*inv_sg),31.f)*s_g), 1e-8f)/31.f;
    const float inv_si=1.f/s_i, inv_sf=1.f/s_f, inv_so=1.f/s_o, inv_sgt=1.f/s_gt;

    float zv[10], ov[10];
    float zmax = 0.f;
    #pragma unroll
    for(int q = 0; q < 10; q++){
      const int e = lane + q*32;
      const int m = e / NHID, j = e % NHID;
      const float gi = fminf(fmaxf(rintf(sh_g[wv][m][j        ]*inv_sg),-32.f),31.f)*s_g;
      const float gf = fminf(fmaxf(rintf(sh_g[wv][m][j+NHID   ]*inv_sg),-32.f),31.f)*s_g;
      const float gg = fminf(fmaxf(rintf(sh_g[wv][m][j+2*NHID ]*inv_sg),-32.f),31.f)*s_g;
      const float go = fminf(fmaxf(rintf(sh_g[wv][m][j+3*NHID ]*inv_sg),-32.f),31.f)*s_g;
      const float iv = fminf(fmaxf(rintf(sigm(gi)*inv_si), 0.f),63.f)*s_i;
      const float fv = fminf(fmaxf(rintf(sigm(gf)*inv_sf), 0.f),63.f)*s_f;
      const float gv = fminf(fmaxf(rintf(tanhf(gg)*inv_sgt),-32.f),31.f)*s_gt;
      const float o_ = fminf(fmaxf(rintf(sigm(go)*inv_so), 0.f),63.f)*s_o;
      const float z  = fv * sh_c[lr0 + m][j] + iv * gv;
      zv[q] = z; ov[q] = o_;
      zmax = fmaxf(zmax, fabsf(z));
    }
    zmax = wave_max(zmax);
    if(lane == 0) atomicMax(&slots[5], f2key(zmax));
    grid_barrier(wsu + WS_CNT, wsu + WS_GEN, nb);
    if(blockIdx.x == 0 && threadIdx.x == 0) oslots[5] = 0u;
    const float Zm  = key2f(((volatile unsigned*)slots)[5]);
    const float s_c = fmaxf(Zm, 1e-8f)/31.f;             const float inv_sc = 1.f/s_c;
    const float s_t = fmaxf(tanhf(fmaxf(Zm,1e-8f)), 1e-8f)/31.f;
    const float inv_st = 1.f/s_t;

    float hmax = 0.f, homax = -3.4e38f;
    float hv[10];
    #pragma unroll
    for(int q = 0; q < 10; q++){
      const int e = lane + q*32;
      const int m = e / NHID, j = e % NHID;
      const float cn = fminf(fmaxf(rintf(zv[q]*inv_sc),-32.f),31.f)*s_c;
      sh_c[lr0 + m][j] = cn;
      const float tq = fminf(fmaxf(rintf(tanhf(cn)*inv_st),-32.f),31.f)*s_t;
      const float ot = ov[q] * tq;
      hv[q] = ot;
      hmax  = fmaxf(hmax, fabsf(ot));
      homax = fmaxf(homax, ot);
    }
    hmax = wave_max(hmax); homax = wave_max(homax);
    if(lane == 0){
      atomicMax(&slots[6], f2key(hmax));
      atomicMax(&slots[7], f2key(homax));
    }
    grid_barrier(wsu + WS_CNT, wsu + WS_GEN, nb);
    if(blockIdx.x == 0 && threadIdx.x == 0){ oslots[6]=0u; oslots[7]=0u; }
    const float Hm = key2f(((volatile unsigned*)slots)[6]);
    homax_g = key2f(((volatile unsigned*)slots)[7]);
    s_h = fmaxf(Hm, 1e-8f)/127.f; inv_sh = 1.f/s_h;
    #pragma unroll
    for(int q = 0; q < 10; q++){
      const int e = lane + q*32;
      const int m = e / NHID, j = e % NHID;
      sh_h[lr0 + m][j] = fminf(fmaxf(rintf(hv[q]*inv_sh),-128.f),127.f)*s_h;
    }
  }

  #pragma unroll
  for(int q = 0; q < 10; q++){
    const int e = lane + q*32;
    const int m = e / NHID, j = e % NHID;
    hn[(rb + m)*NHID + j] = sh_h[lr0 + m][j];
  }
  if(blockIdx.x == 0 && threadIdx.x == 0){
    const float mh = fminf(fmaxf(rintf(homax_g*inv_sh),-128.f),127.f)*s_h;
    wsu[WS_MAXHN] = f2key(mh);
  }
}

template<int K, int N>
__global__ __launch_bounds__(256) void mlp_layer(
    const float* __restrict__ in, const float* __restrict__ W, const float* __restrict__ bias,
    const unsigned* __restrict__ in_maxkey, float* __restrict__ out, unsigned* out_maxkey, int B)
{
  const int  wv = threadIdx.x >> 5, lane = threadIdx.x & 31;
  const int  mrow = lane & 15, hi = lane >> 4;
  const long rb = ((long)blockIdx.x * WAVES_PER_BLOCK + wv) * 16;
  if(rb >= B) return;

  float wm = 0.f;
  for(int i = lane; i < N*K; i += 32) wm = fmaxf(wm, fabsf(W[i]));
  const float s_w = fmaxf(wave_max(wm), 1e-8f)/127.f;
  const float inv_sw = 1.f/s_w;
  const float s_in = fmaxf(key2f(*(volatile const unsigned*)in_maxkey), 1e-8f)/63.f;
  const float inv_sin = 1.f/s_in;

  float omax = -3.4e38f;
  #pragma unroll
  for(int nt = 0; nt < (N+15)/16; nt++){
    const int n = nt*16 + mrow;
    v8f acc = {0.f,0.f,0.f,0.f,0.f,0.f,0.f,0.f};
    #pragma unroll
    for(int kt = 0; kt < (K+3)/4; kt++){
      v2f A, Bm;
      #pragma unroll
      for(int v = 0; v < 2; v++){
        const int k = kt*4 + v + hi*2;
        float a = 0.f, bw = 0.f;
        if(k < K){
          a = fminf(fmaxf(rintf(in[(rb + mrow)*K + k]*inv_sin), 0.f), 63.f)*s_in;
          if(n < N) bw = fminf(fmaxf(rintf(W[n*K + k]*inv_sw), -128.f), 127.f)*s_w;
        }
        A[v] = a; Bm[v] = bw;
      }
      acc = __builtin_amdgcn_wmma_f32_16x16x4_f32(false, A, false, Bm, (short)0, acc, false, false);
    }
    if(n < N){
      const float bb = bias[n];
      #pragma unroll
      for(int r = 0; r < 8; r++){
        const float o = acc[r] + bb;
        out[(rb + r + hi*8)*N + n] = o;
        omax = fmaxf(omax, o);
      }
    }
  }
  if(out_maxkey){
    omax = wave_max(omax);
    if(lane == 0) atomicMax(out_maxkey, f2key(omax));
  }
}

extern "C" void kernel_launch(void* const* d_in, const int* in_sizes, int n_in,
                              void* d_out, int out_size, void* d_ws, size_t ws_size,
                              hipStream_t stream){
  (void)n_in; (void)out_size; (void)ws_size;
  const float* x    = (const float*)d_in[0];
  const float* w_ih = (const float*)d_in[1];
  const float* w_hh = (const float*)d_in[2];
  const float* b    = (const float*)d_in[3];
  const float* W1   = (const float*)d_in[4];
  const float* b1   = (const float*)d_in[5];
  const float* W2   = (const float*)d_in[6];
  const float* b2   = (const float*)d_in[7];
  const float* W3   = (const float*)d_in[8];
  const float* b3   = (const float*)d_in[9];
  const int B = in_sizes[0] / (T_STEPS * NIN);

  unsigned* wsu = (unsigned*)d_ws;
  float*    wsf = (float*)d_ws;
  float* hn = wsf + WS_HN;
  float* a1 = hn + (size_t)B * NHID;
  float* a2 = a1 + (size_t)B * 64;

  const int nb = B / (16 * WAVES_PER_BLOCK);

  init_ws<<<1, 64, 0, stream>>>(wsu);
  xmax_kernel<<<T_STEPS, 256, 0, stream>>>(x, wsf + WS_XMAX, B);
  lstm_scan<<<nb, 256, 0, stream>>>(x, w_ih, w_hh, b, wsf, wsu, hn, B, nb);
  mlp_layer<NHID, 64><<<nb, 256, 0, stream>>>(hn, W1, b1, wsu + WS_MAXHN, a1, wsu + WS_MAXA1, B);
  mlp_layer<64, 32><<<nb, 256, 0, stream>>>(a1, W2, b2, wsu + WS_MAXA1, a2, wsu + WS_MAXA2, B);
  mlp_layer<32, 5><<<nb, 256, 0, stream>>>(a2, W3, b3, wsu + WS_MAXA2, (float*)d_out, nullptr, B);
}
